// Phi3Attention_77704548319817
// MI455X (gfx1250) — matrix operand order repaired and checked by reading, compiled, not run on hardware
//
#include <hip/hip_runtime.h>
#include <hip/hip_bf16.h>
#include <math.h>
#include <stdint.h>

typedef __bf16 bf16_t;
typedef __attribute__((ext_vector_type(8)))  __bf16 bf16x8;
typedef __attribute__((ext_vector_type(16))) __bf16 bf16x16;
typedef __attribute__((ext_vector_type(8)))  float  floatx8;
typedef __attribute__((ext_vector_type(4)))  unsigned int uint32x4;
typedef __attribute__((ext_vector_type(8)))  int int32x8;
typedef __attribute__((ext_vector_type(4)))  int int32x4;

#define B_   2
#define S_   2048
#define H_   3072
#define NH_  32
#define HD_  96
#define O_   9216          /* NH*HD + 2*NKV*HD */
#define NEGV   (-1.0e9f)
#define RSCALE 0.10206207261596575f   /* 1/sqrt(96) */

static __device__ inline bf16x16 cat8(bf16x8 lo, bf16x8 hi) {
    bf16x16 r;
#pragma unroll
    for (int i = 0; i < 8; ++i) { r[i] = lo[i]; r[i + 8] = hi[i]; }
    return r;
}

static __device__ inline unsigned short f2bf_bits(float f) {
    bf16_t h = (bf16_t)f;
    return __builtin_bit_cast(unsigned short, h);
}

/* Issue a TDM load of a 2D bf16 tile: 32 rows x 64 cols from a row-major
 * [M x K] matrix (row stride K elems), tile origin = gaddr, into LDS at
 * byte offset lds_off.  D# layout per CDNA5 ISA ch.8 (group0/group1). */
static __device__ inline void tdm_load_a32x64(unsigned lds_off, const bf16_t* gaddr,
                                              int K, int M) {
    unsigned long long ga = (unsigned long long)(uintptr_t)gaddr;
    uint32x4 g0;
    g0[0] = 1u;                                          /* count=1, user desc    */
    g0[1] = lds_off;                                     /* lds_addr (bytes)      */
    g0[2] = (unsigned)(ga & 0xFFFFFFFFull);              /* global_addr[31:0]     */
    g0[3] = (unsigned)((ga >> 32) & 0x1FFFFFFull) | (2u << 30); /* addr[56:32], type=2 */

    int32x8 g1;
    g1[0] = (1 << 16);                                   /* wg_mask=0, data_size=2B */
    g1[1] = (K & 0xFFFF) << 16;                          /* tensor_dim0[15:0]     */
    g1[2] = ((K >> 16) & 0xFFFF) | ((M & 0xFFFF) << 16); /* dim0[31:16], dim1[15:0] */
    g1[3] = ((M >> 16) & 0xFFFF) | (64 << 16);           /* dim1[31:16], tile_dim0=64 */
    g1[4] = 32;                                          /* tile_dim1=32, tile_dim2=0 */
    g1[5] = K;                                           /* tensor_dim0_stride[31:0] */
    g1[6] = 0;
    g1[7] = 0;
    int32x4 z4 = {0, 0, 0, 0};                           /* 2D tensor: groups 2/3 unused */
    int32x8 z8 = {0, 0, 0, 0, 0, 0, 0, 0};
    __builtin_amdgcn_tensor_load_to_lds(g0, g1, z4, z4, z8, 0);
}

/* ---------------- fp32 -> bf16 conversion ---------------- */
__global__ __launch_bounds__(256) void cvt_f32_bf16(const float* __restrict__ in,
                                                    bf16_t* __restrict__ out, int n) {
    int i = blockIdx.x * blockDim.x + threadIdx.x;
    if (i < n) out[i] = (bf16_t)in[i];
}

/* ---------------- C = A(MxK) * W(NxK)^T, bf16 WMMA + TDM-staged A -------------
 * Block = 8 waves = 2 (M) x 4 (N) -> 32x256 block tile.  A block tile
 * (32 rows x 64 K, 4KB) is DMA'd into LDS by the Tensor Data Mover,
 * double-buffered and overlapped with WMMA; B streams via global_load_b128.
 */
__global__ __launch_bounds__(256) void gemm_bf16_wmma(const bf16_t* __restrict__ A,
                                                      const bf16_t* __restrict__ W,
                                                      float* __restrict__ outF,
                                                      bf16_t* __restrict__ outB,
                                                      int M, int N, int K) {
    __shared__ unsigned short ldsA[2 * 32 * 64];   /* 2 x 4KB, at LDS offset 0 */

    const int NT = 4;
    int widx = threadIdx.x >> 5;
    int lane = threadIdx.x & 31;
    int l16  = lane & 15;
    int half = lane >> 4;
    int mg   = widx >> 2;          /* 0..1: which 16-row strip   */
    int ng   = widx & 3;           /* 0..3: which 64-col group   */

    int blocksN = N / 256;
    int bm = blockIdx.x / blocksN;
    int bn = blockIdx.x % blocksN;

    int rowBase = bm * 32;                    /* block's first A row        */
    int colBase = bn * 256 + ng * 64;         /* this wave's first out col  */

    floatx8 acc[NT];
#pragma unroll
    for (int t = 0; t < NT; ++t) acc[t] = (floatx8){0,0,0,0,0,0,0,0};

    int nch = K / 64;
    if (widx == 0)
        tdm_load_a32x64(0u, A + (size_t)rowBase * K, K, M);

    for (int kc = 0; kc < nch; ++kc) {
        if (widx == 0) {
            if (kc + 1 < nch) {
                tdm_load_a32x64((unsigned)(((kc + 1) & 1) * 4096),
                                A + (size_t)rowBase * K + (kc + 1) * 64, K, M);
                __builtin_amdgcn_s_wait_tensorcnt(1);   /* tile kc complete */
            } else {
                __builtin_amdgcn_s_wait_tensorcnt(0);
            }
        }
        __syncthreads();                                /* tile kc visible  */

        const unsigned short* abuf = ldsA + (kc & 1) * (32 * 64);
#pragma unroll
        for (int sub = 0; sub < 64; sub += 32) {
            const unsigned short* arow = abuf + (mg * 16 + l16) * 64 + sub;
            bf16x16 av = *(const bf16x16*)(arow + 16 * half);
#pragma unroll
            for (int t = 0; t < NT; ++t) {
                int col = colBase + t * 16 + l16;
                bf16x16 bv = *(const bf16x16*)(W + (size_t)col * K + kc * 64 + sub + 16 * half);
                acc[t] = __builtin_amdgcn_wmma_f32_16x16x32_bf16(
                    false, av, false, bv, (short)0, acc[t], false, false);
            }
        }
        __syncthreads();                     /* done reading before overwrite */
    }

#pragma unroll
    for (int t = 0; t < NT; ++t) {
        int col = colBase + t * 16 + l16;
#pragma unroll
        for (int v = 0; v < 8; ++v) {
            int row = rowBase + mg * 16 + v + 8 * half;
            if (outF) outF[(size_t)row * N + col] = acc[t][v];
            else      outB[(size_t)row * N + col] = (bf16_t)acc[t][v];
        }
    }
}

/* ---------------- RoPE + layout: qkv[4096,9216] -> Q,K [B,NH,S,HD], Vt [B,NH,HD,S] */
__global__ __launch_bounds__(256) void rope_kernel(const bf16_t* __restrict__ qkv,
                                                   bf16_t* __restrict__ Q,
                                                   bf16_t* __restrict__ K,
                                                   bf16_t* __restrict__ Vt) {
    int idx = blockIdx.x * blockDim.x + threadIdx.x;
    int total = B_ * NH_ * S_ * HD_;
    if (idx >= total) return;
    int d = idx % HD_;
    int t = idx / HD_;
    int s = t % S_;
    int t2 = t / S_;
    int h = t2 % NH_;
    int b = t2 / NH_;

    const bf16_t* base = qkv + (size_t)(b * S_ + s) * O_;
    float qv = (float)base[h * HD_ + d];
    float kv = (float)base[NH_ * HD_ + h * HD_ + d];
    float vv = (float)base[2 * NH_ * HD_ + h * HD_ + d];

    int dm = d % 48;
    float invf = __expf(-0.19188208945779527f * (float)dm);  /* 10000^(-2*dm/96) */
    float ang = (float)s * invf;
    float c = __cosf(ang), sn = __sinf(ang);

    float qr = (d < 48) ? -(float)base[h * HD_ + d + 48]
                        :  (float)base[h * HD_ + d - 48];
    float kr = (d < 48) ? -(float)base[NH_ * HD_ + h * HD_ + d + 48]
                        :  (float)base[NH_ * HD_ + h * HD_ + d - 48];

    size_t qi = ((size_t)(b * NH_ + h) * S_ + s) * HD_ + d;
    Q[qi] = (bf16_t)(qv * c + qr * sn);
    K[qi] = (bf16_t)(kv * c + kr * sn);
    Vt[((size_t)(b * NH_ + h) * HD_ + d) * S_ + s] = (bf16_t)vv;
}

/* ---------------- causal flash attention, one wave per 16-query block -------- */
__global__ __launch_bounds__(128) void attn_kernel(const bf16_t* __restrict__ Q,
                                                   const bf16_t* __restrict__ K,
                                                   const bf16_t* __restrict__ Vt,
                                                   bf16_t* __restrict__ Obf) {
    __shared__ unsigned short ldsP[4][16 * 32];

    int widx = threadIdx.x >> 5;
    int lane = threadIdx.x & 31;
    int l16  = lane & 15;
    int half = lane >> 4;

    int task = blockIdx.x * 4 + widx;
    int total = B_ * NH_ * (S_ / 16);
    if (task >= total) return;

    int qb  = task % (S_ / 16);
    int t2  = task / (S_ / 16);
    int h   = t2 % NH_;
    int b   = t2 / NH_;
    int qbase = qb * 16;

    const bf16_t* Qp  = Q  + ((size_t)(b * NH_ + h) * S_ + qbase) * HD_;
    const bf16_t* Kp  = K  + ((size_t)(b * NH_ + h) * S_) * HD_;
    const bf16_t* Vtp = Vt + ((size_t)(b * NH_ + h) * HD_) * S_;

    /* Q in A-matrix layout for 3 K-chunks of 32 over HD=96 */
    bf16x16 qa[3];
#pragma unroll
    for (int c = 0; c < 3; ++c) {
        const bf16_t* qrow = Qp + (size_t)l16 * HD_ + c * 32;
        qa[c] = *(const bf16x16*)(qrow + 16 * half);
    }

    floatx8 acc[6];
#pragma unroll
    for (int t = 0; t < 6; ++t) acc[t] = (floatx8){0,0,0,0,0,0,0,0};
    float mrow[8], lrow[8];
#pragma unroll
    for (int v = 0; v < 8; ++v) { mrow[v] = -3.0e38f; lrow[v] = 0.0f; }

    for (int kb = 0; kb < qbase + 16; kb += 32) {
        bool haveT1 = (kb < qbase);   /* wave-uniform */

        floatx8 s0 = (floatx8){0,0,0,0,0,0,0,0};
        floatx8 s1 = (floatx8){0,0,0,0,0,0,0,0};
#pragma unroll
        for (int c = 0; c < 3; ++c) {
            int dbase = c * 32 + 16 * half;
            bf16x16 b0 = *(const bf16x16*)(Kp + (size_t)(kb + l16) * HD_ + dbase);
            s0 = __builtin_amdgcn_wmma_f32_16x16x32_bf16(
                false, qa[c], false, b0, (short)0, s0, false, false);
            if (haveT1) {
                bf16x16 b1 = *(const bf16x16*)(Kp + (size_t)(kb + 16 + l16) * HD_ + dbase);
                s1 = __builtin_amdgcn_wmma_f32_16x16x32_bf16(
                    false, qa[c], false, b1, (short)0, s1, false, false);
            }
        }

        /* scale + causal mask + online softmax (16-lane half-wave rows) */
#pragma unroll
        for (int v = 0; v < 8; ++v) {
            int qrow = qbase + v + 8 * half;
            float x0 = s0[v] * RSCALE + (((kb + l16) > qrow) ? NEGV : 0.0f);
            float x1 = haveT1 ? (s1[v] * RSCALE + (((kb + 16 + l16) > qrow) ? NEGV : 0.0f))
                              : -3.0e38f;
            float xm = fmaxf(x0, x1);
#pragma unroll
            for (int o = 8; o >= 1; o >>= 1) xm = fmaxf(xm, __shfl_xor(xm, o, 32));
            float mnew = fmaxf(mrow[v], xm);
            float sc = __expf(mrow[v] - mnew);
            float p0 = __expf(x0 - mnew);
            float p1 = haveT1 ? __expf(x1 - mnew) : 0.0f;
            float rs = p0 + p1;
#pragma unroll
            for (int o = 8; o >= 1; o >>= 1) rs += __shfl_xor(rs, o, 32);
            lrow[v] = lrow[v] * sc + rs;
            mrow[v] = mnew;
#pragma unroll
            for (int dt = 0; dt < 6; ++dt) acc[dt][v] *= sc;

            int r = v + 8 * half;
            ldsP[widx][r * 32 + l16]      = f2bf_bits(p0);
            ldsP[widx][r * 32 + 16 + l16] = f2bf_bits(p1);
        }

        __asm__ volatile("s_wait_dscnt 0" ::: "memory");

        /* P (C layout) -> A layout via LDS */
        const unsigned short* prow = &ldsP[widx][l16 * 32];
        bf16x16 pv = *(const bf16x16*)(prow + 16 * half);

        int keyb = (haveT1 || half == 0) ? (kb + 16 * half) : kb;  /* avoid OOB; P cols are 0 there */
#pragma unroll
        for (int dt = 0; dt < 6; ++dt) {
            int d = dt * 16 + l16;
            bf16x16 bv = *(const bf16x16*)(Vtp + (size_t)d * S_ + keyb);
            acc[dt] = __builtin_amdgcn_wmma_f32_16x16x32_bf16(
                false, pv, false, bv, (short)0, acc[dt], false, false);
        }
        __asm__ volatile("" ::: "memory");  /* keep LDS reuse ordered across iterations */
    }

    /* normalize + store bf16 into [B, S, NH*HD] */
#pragma unroll
    for (int v = 0; v < 8; ++v) {
        float invl = 1.0f / lrow[v];
        int srow = qbase + v + 8 * half;
        size_t rowoff = (size_t)(b * S_ + srow) * H_ + h * HD_;
#pragma unroll
        for (int dt = 0; dt < 6; ++dt)
            Obf[rowoff + dt * 16 + l16] = (bf16_t)(acc[dt][v] * invl);
    }
}

/* ---------------- host side ---------------- */
extern "C" void kernel_launch(void* const* d_in, const int* in_sizes, int n_in,
                              void* d_out, int out_size, void* d_ws, size_t ws_size,
                              hipStream_t stream) {
    const float* hidden = (const float*)d_in[0];
    /* d_in[1] position_ids, d_in[2] attention_mask: causal+arange reproduced in-kernel */
    const float* w_qkv  = (const float*)d_in[3];
    const float* w_o    = (const float*)d_in[4];
    float* out = (float*)d_out;

    char* ws = (char*)d_ws;
    const size_t nX    = (size_t)B_ * S_ * H_;        /* 12.6M  */
    const size_t nWqkv = (size_t)O_ * H_;             /* 28.3M  */
    const size_t nWo   = (size_t)H_ * H_;             /* 9.4M   */
    const size_t nQKV  = (size_t)B_ * S_ * O_;        /* 37.7M  */
    const size_t nHead = (size_t)B_ * NH_ * S_ * HD_; /* 12.6M  */

    bf16_t* Xbf   = (bf16_t*)(ws);
    bf16_t* Wqkvb = (bf16_t*)(ws + 2 * nX);
    bf16_t* Wob   = (bf16_t*)(ws + 2 * (nX + nWqkv));
    bf16_t* QKVb  = (bf16_t*)(ws + 2 * (nX + nWqkv + nWo));
    bf16_t* Qb    = (bf16_t*)(ws + 2 * (nX + nWqkv + nWo + nQKV));
    bf16_t* Kb    = (bf16_t*)(ws + 2 * (nX + nWqkv + nWo + nQKV + nHead));
    bf16_t* Vtb   = (bf16_t*)(ws + 2 * (nX + nWqkv + nWo + nQKV + 2 * nHead));
    bf16_t* AOb   = Xbf;  /* X no longer needed after GEMM1: reuse for attention output */

    cvt_f32_bf16<<<(int)((nX    + 255) / 256), 256, 0, stream>>>(hidden, Xbf,   (int)nX);
    cvt_f32_bf16<<<(int)((nWqkv + 255) / 256), 256, 0, stream>>>(w_qkv,  Wqkvb, (int)nWqkv);
    cvt_f32_bf16<<<(int)((nWo   + 255) / 256), 256, 0, stream>>>(w_o,    Wob,   (int)nWo);

    /* GEMM1: qkv[4096, 9216] = X[4096,3072] * Wqkv[9216,3072]^T  (bf16 out) */
    {
        int M = B_ * S_, N = O_, K = H_;
        int blocks = (M / 32) * (N / 256);
        gemm_bf16_wmma<<<blocks, 256, 0, stream>>>(Xbf, Wqkvb, nullptr, QKVb, M, N, K);
    }

    rope_kernel<<<(int)((nHead + 255) / 256), 256, 0, stream>>>(QKVb, Qb, Kb, Vtb);

    attn_kernel<<<(B_ * NH_ * (S_ / 16)) / 4, 128, 0, stream>>>(Qb, Kb, Vtb, AOb);

    /* GEMM2: out[4096, 3072] = AO[4096,3072] * Wo[3072,3072]^T  (f32 out) */
    {
        int M = B_ * S_, N = H_, K = H_;
        int blocks = (M / 32) * (N / 256);
        gemm_bf16_wmma<<<blocks, 256, 0, stream>>>(AOb, Wob, out, nullptr, M, N, K);
    }
}
